// MultiHeadAttention_2980707303646
// MI455X (gfx1250) — matrix operand order repaired and checked by reading, not compiled, not run on hardware
//
#include <hip/hip_runtime.h>

// ---------------------------------------------------------------------------
// MHA for MI455X (gfx1250).
//   0) prep: x -> bf16 (Xb), W_qkv / W_o -> bf16 transposed
//   1) QKV GEMM: 4-wave workgroups, 64x64 macro-tile (2x2 WMMAs per wave per
//      k-step), double-buffered LDS slabs via GLOBAL_LOAD_ASYNC_TO_LDS_B128
//   2) flash attention (causal+pad), online softmax, LDS P re-layout
//   3) output projection GEMM (same async-LDS scheme) -> fp32
// All matrix math on v_wmma_f32_16x16x32_bf16.
// ---------------------------------------------------------------------------

#define S_LEN   4096
#define DM      1024
#define NH      16
#define HD      64
#define DQKV    3072

typedef __attribute__((ext_vector_type(16))) __bf16 v16bf;
typedef __attribute__((ext_vector_type(8)))  float  v8f;
typedef int v4i_vec __attribute__((vector_size(16)));   // matches builtin proto

#if __has_builtin(__builtin_amdgcn_global_load_async_to_lds_b128) && \
    __has_builtin(__builtin_amdgcn_s_wait_asynccnt)
#define HAVE_ASYNC 1
#else
#define HAVE_ASYNC 0
#endif

union FragU {
  v16bf bf;
  unsigned short us[16];
  uint4 q[2];
};

__device__ __forceinline__ unsigned short f2bf(float f) {
  unsigned u = __builtin_bit_cast(unsigned, f);
  u += 0x7FFFu + ((u >> 16) & 1u);          // round-to-nearest-even
  return (unsigned short)(u >> 16);
}

__device__ __forceinline__ v8f wmma_bf16(FragU a, FragU b, v8f c) {
  return __builtin_amdgcn_wmma_f32_16x16x32_bf16(false, a.bf, false, b.bf,
                                                 (short)0, c, false, false);
}

// one 16-byte async (or fallback sync) copy global -> LDS
__device__ __forceinline__ void copy16_to_lds(const unsigned short* g,
                                              unsigned short* l)
{
#if HAVE_ASYNC
  __builtin_amdgcn_global_load_async_to_lds_b128(
      (__attribute__((address_space(1))) v4i_vec*)g,
      (__attribute__((address_space(3))) v4i_vec*)l, 0, 0);
#else
  *reinterpret_cast<uint4*>(l) = *reinterpret_cast<const uint4*>(g);
#endif
}

#if HAVE_ASYNC
#define WAIT_ASYNC(n) __builtin_amdgcn_s_wait_asynccnt(n)
#else
#define WAIT_ASYNC(n)
#endif

// ---------------------------------------------------------------------------
// Prep 0a: elementwise fp32 -> bf16
// ---------------------------------------------------------------------------
__global__ __launch_bounds__(256)
void cvt_kernel(const float* __restrict__ in, unsigned short* __restrict__ out)
{
  const size_t i = (size_t)blockIdx.x * 256 + threadIdx.x;   // float4 index
  const float4 f = reinterpret_cast<const float4*>(in)[i];
  uint2 p;
  p.x = (unsigned)f2bf(f.x) | ((unsigned)f2bf(f.y) << 16);
  p.y = (unsigned)f2bf(f.z) | ((unsigned)f2bf(f.w) << 16);
  reinterpret_cast<uint2*>(out)[i] = p;
}

// ---------------------------------------------------------------------------
// Prep 0b: fp32 [R][C] -> bf16 transposed [C][R], 32x32 LDS tiles.
// ---------------------------------------------------------------------------
__global__ __launch_bounds__(256)
void trans_kernel(const float* __restrict__ in, unsigned short* __restrict__ out,
                  int R, int C)
{
  __shared__ float tile[32][33];
  const int tx = threadIdx.x, ty = threadIdx.y;
  const int c0 = blockIdx.x * 32, r0 = blockIdx.y * 32;
#pragma unroll
  for (int j = 0; j < 32; j += 8)
    tile[ty + j][tx] = in[(size_t)(r0 + ty + j) * C + c0 + tx];
  __syncthreads();
#pragma unroll
  for (int j = 0; j < 32; j += 8)
    out[(size_t)(c0 + ty + j) * R + r0 + tx] = f2bf(tile[tx][ty + j]);
}

// ---------------------------------------------------------------------------
// Macro-tile GEMM core: 128 threads (4 waves), 64x64 output tile,
// K-loop of 32, double-buffered 64x32 LDS slabs, async staging.
// Wave w -> 32x32 subtile at (wr=w>>1, wc=w&1); acc[rt][ct] = its 2x2 tiles.
// A: row-major [*, DM] bf16;  B: row-major-of-transpose [col][DM] bf16.
// ---------------------------------------------------------------------------
__device__ __forceinline__ void gemm_macro(const unsigned short* __restrict__ Ag,
                                           const unsigned short* __restrict__ Bg,
                                           int row0, int col0,
                                           unsigned short (*As)[64 * 32],
                                           unsigned short (*Bs)[64 * 32],
                                           v8f acc[2][2])
{
  const int t    = threadIdx.x;       // 0..127
  const int lane = t & 31;
  const int w    = t >> 5;            // wave 0..3
  const int wr   = w >> 1, wc = w & 1;
  const int n    = lane & 15;
  const int half = lane >> 4;
  int cur = 0;

  // staging: each thread moves 2 x 16B of A and 2 x 16B of B per k-step
  auto stage = [&](int buf, int k0) {
#pragma unroll
    for (int c = 0; c < 2; ++c) {
      const int ch   = c * 128 + t;   // 0..255 chunk id
      const int srow = ch >> 2;       // 0..63
      const int skq  = (ch & 3) * 8;  // k offset in shorts
      copy16_to_lds(Ag + (size_t)(row0 + srow) * DM + k0 + skq,
                    &As[buf][srow * 32 + skq]);
      copy16_to_lds(Bg + (size_t)(col0 + srow) * DM + k0 + skq,
                    &Bs[buf][srow * 32 + skq]);
    }
  };

  stage(0, 0);
  for (int k0 = 0; k0 < DM; k0 += 32) {
    if (k0 + 32 < DM) {
      stage(cur ^ 1, k0 + 32);
      WAIT_ASYNC(4);                  // current slab done; next in flight
    } else {
      WAIT_ASYNC(0);
    }
    __syncthreads();

    FragU a[2], b[2];
#pragma unroll
    for (int rt = 0; rt < 2; ++rt) {
      const unsigned short* la = &As[cur][(wr * 32 + rt * 16 + n) * 32 + half * 8];
      a[rt].q[0] = *reinterpret_cast<const uint4*>(la);
      a[rt].q[1] = *reinterpret_cast<const uint4*>(la + 16);
    }
#pragma unroll
    for (int ct = 0; ct < 2; ++ct) {
      const unsigned short* lb = &Bs[cur][(wc * 32 + ct * 16 + n) * 32 + half * 8];
      b[ct].q[0] = *reinterpret_cast<const uint4*>(lb);
      b[ct].q[1] = *reinterpret_cast<const uint4*>(lb + 16);
    }
#pragma unroll
    for (int rt = 0; rt < 2; ++rt)
#pragma unroll
      for (int ct = 0; ct < 2; ++ct)
        acc[rt][ct] = wmma_bf16(a[rt], b[ct], acc[rt][ct]);

    __syncthreads();                  // all reads done before buffer reuse
    cur ^= 1;
  }
}

// ---------------------------------------------------------------------------
// Kernel 1: qkv = x @ W_qkv + b_qkv.  grid = (DQKV/64, S/64), block = 128.
// Q,K row-major [s][1024]; V transposed [1024][s].
// ---------------------------------------------------------------------------
__global__ __launch_bounds__(128)
void qkv_kernel(const unsigned short* __restrict__ Xb,
                const unsigned short* __restrict__ Wqkvt,   // [3072][1024]
                const float* __restrict__ bqkv,
                unsigned short* __restrict__ Qb,
                unsigned short* __restrict__ Kb,
                unsigned short* __restrict__ Vt)
{
  __shared__ unsigned short As[2][64 * 32];
  __shared__ unsigned short Bs[2][64 * 32];

  const int lane = threadIdx.x & 31;
  const int w    = threadIdx.x >> 5;
  const int wr   = w >> 1, wc = w & 1;
  const int n    = lane & 15;
  const int half = lane >> 4;
  const int row0 = blockIdx.y * 64;
  const int col0 = blockIdx.x * 64;

  v8f acc[2][2] = {};
  gemm_macro(Xb, Wqkvt, row0, col0, As, Bs, acc);

#pragma unroll
  for (int ct = 0; ct < 2; ++ct) {
    const int col  = col0 + wc * 32 + ct * 16 + n;
    const float bias = bqkv[col];
    const int   buf  = col >> 10;            // 0=Q 1=K 2=V
    const int   cc   = col & 1023;
#pragma unroll
    for (int rt = 0; rt < 2; ++rt)
#pragma unroll
      for (int r = 0; r < 8; ++r) {
        const int row = row0 + wr * 32 + rt * 16 + r + 8 * half;
        const unsigned short v = f2bf(acc[rt][ct][r] + bias);
        if (buf == 0)      Qb[(size_t)row * DM + cc] = v;
        else if (buf == 1) Kb[(size_t)row * DM + cc] = v;
        else               Vt[(size_t)cc * S_LEN + row] = v;
      }
  }
}

// ---------------------------------------------------------------------------
// Kernel 2: flash attention, causal + padding mask.
// grid = (256 query-tiles, 16 heads), block = 1 wave.
// ---------------------------------------------------------------------------
__global__ __launch_bounds__(32)
void attn_kernel(const unsigned short* __restrict__ Q,
                 const unsigned short* __restrict__ K,
                 const unsigned short* __restrict__ Vt,    // [1024][s]
                 const unsigned char*  __restrict__ pad,
                 unsigned short* __restrict__ Cb)
{
  __shared__ unsigned short ldsP[16 * 32];

  const int lane  = threadIdx.x;
  const int n     = lane & 15;
  const int half  = lane >> 4;
  const int qi    = blockIdx.x;
  const int h     = blockIdx.y;
  const int hb    = h * HD;
  const int qrow0 = qi * 16;

  v8f   o[4] = {};
  float mrow[8], lrow[8];
#pragma unroll
  for (int r = 0; r < 8; ++r) { mrow[r] = -1e30f; lrow[r] = 0.0f; }

  FragU qa[2];
#pragma unroll
  for (int c = 0; c < 2; ++c) {
    const unsigned short* p = Q + (size_t)(qrow0 + n) * DM + hb + c * 32 + half * 8;
    qa[c].q[0] = *reinterpret_cast<const uint4*>(p);
    qa[c].q[1] = *reinterpret_cast<const uint4*>(p + 16);
  }

  const int nkb = (qrow0 + 16 + 31) >> 5;
  for (int bblk = 0; bblk < nkb; ++bblk) {
    const int kb = bblk * 32;

    if (bblk + 1 < nkb) {
      __builtin_prefetch(K  + (size_t)(kb + 32 + n) * DM + hb, 0, 3);
      __builtin_prefetch(Vt + (size_t)(hb + n) * S_LEN + kb + 32, 0, 3);
    }

    v8f s0 = {}, s1 = {};
#pragma unroll
    for (int c = 0; c < 2; ++c) {
      FragU kf0, kf1;
      const unsigned short* kp0 = K + (size_t)(kb + n) * DM + hb + c * 32 + half * 8;
      kf0.q[0] = *reinterpret_cast<const uint4*>(kp0);
      kf0.q[1] = *reinterpret_cast<const uint4*>(kp0 + 16);
      const unsigned short* kp1 = kp0 + (size_t)16 * DM;
      kf1.q[0] = *reinterpret_cast<const uint4*>(kp1);
      kf1.q[1] = *reinterpret_cast<const uint4*>(kp1 + 16);
      s0 = wmma_bf16(qa[c], kf0, s0);
      s1 = wmma_bf16(qa[c], kf1, s1);
    }

    float corr[8];
#pragma unroll
    for (int r = 0; r < 8; ++r) {
      const int qrow = qrow0 + r + 8 * half;
      const int kc0  = kb + n;
      const int kc1  = kb + 16 + n;
      float v0 = (kc0 <= qrow && pad[kc0]) ? s0[r] * 0.125f : -1e30f;
      float v1 = (kc1 <= qrow && pad[kc1]) ? s1[r] * 0.125f : -1e30f;
      float mx = fmaxf(v0, v1);
#pragma unroll
      for (int off = 1; off < 16; off <<= 1)
        mx = fmaxf(mx, __shfl_xor(mx, off, 16));
      const float nm = fmaxf(mrow[r], mx);
      corr[r] = __expf(mrow[r] - nm);
      const float e0 = __expf(v0 - nm);
      const float e1 = __expf(v1 - nm);
      float rs = e0 + e1;
#pragma unroll
      for (int off = 1; off < 16; off <<= 1)
        rs += __shfl_xor(rs, off, 16);
      lrow[r] = lrow[r] * corr[r] + rs;
      mrow[r] = nm;
      ldsP[(r + 8 * half) * 32 + n]      = f2bf(e0);
      ldsP[(r + 8 * half) * 32 + 16 + n] = f2bf(e1);
    }

#pragma unroll
    for (int nt = 0; nt < 4; ++nt)
#pragma unroll
      for (int r = 0; r < 8; ++r) o[nt][r] *= corr[r];

    FragU pf;
    const unsigned short* lp = ldsP + n * 32 + half * 8;
    pf.q[0] = *reinterpret_cast<const uint4*>(lp);
    pf.q[1] = *reinterpret_cast<const uint4*>(lp + 16);

#pragma unroll
    for (int nt = 0; nt < 4; ++nt) {
      FragU vf;
      const unsigned short* vp = Vt + (size_t)(hb + nt * 16 + n) * S_LEN + kb + half * 8;
      vf.q[0] = *reinterpret_cast<const uint4*>(vp);
      vf.q[1] = *reinterpret_cast<const uint4*>(vp + 16);
      o[nt] = wmma_bf16(pf, vf, o[nt]);
    }
  }

#pragma unroll
  for (int r = 0; r < 8; ++r) {
    const float inv = 1.0f / lrow[r];
    const int   row = qrow0 + r + 8 * half;
#pragma unroll
    for (int nt = 0; nt < 4; ++nt)
      Cb[(size_t)row * DM + hb + nt * 16 + n] = f2bf(o[nt][r] * inv);
  }
}

// ---------------------------------------------------------------------------
// Kernel 3: out = concat @ W_o + b_o.  grid = (DM/64, S/64), block = 128.
// ---------------------------------------------------------------------------
__global__ __launch_bounds__(128)
void proj_kernel(const unsigned short* __restrict__ Cb,
                 const unsigned short* __restrict__ Wot,   // [1024][1024]
                 const float* __restrict__ bo,
                 float* __restrict__ out)
{
  __shared__ unsigned short As[2][64 * 32];
  __shared__ unsigned short Bs[2][64 * 32];

  const int lane = threadIdx.x & 31;
  const int w    = threadIdx.x >> 5;
  const int wr   = w >> 1, wc = w & 1;
  const int n    = lane & 15;
  const int half = lane >> 4;
  const int row0 = blockIdx.y * 64;
  const int col0 = blockIdx.x * 64;

  v8f acc[2][2] = {};
  gemm_macro(Cb, Wot, row0, col0, As, Bs, acc);

#pragma unroll
  for (int ct = 0; ct < 2; ++ct) {
    const int col  = col0 + wc * 32 + ct * 16 + n;
    const float bias = bo[col];
#pragma unroll
    for (int rt = 0; rt < 2; ++rt)
#pragma unroll
      for (int r = 0; r < 8; ++r) {
        const int row = row0 + wr * 32 + rt * 16 + r + 8 * half;
        out[(size_t)row * DM + col] = acc[rt][ct][r] + bias;
      }
  }
}

// ---------------------------------------------------------------------------
extern "C" void kernel_launch(void* const* d_in, const int* in_sizes, int n_in,
                              void* d_out, int out_size, void* d_ws, size_t ws_size,
                              hipStream_t stream)
{
  const float*         x    = (const float*)d_in[0];
  const unsigned char* pad  = (const unsigned char*)d_in[1];
  const float*         Wqkv = (const float*)d_in[2];
  const float*         bqkv = (const float*)d_in[3];
  const float*         Wo   = (const float*)d_in[4];
  const float*         bo   = (const float*)d_in[5];
  float*               out  = (float*)d_out;

  unsigned short* p   = (unsigned short*)d_ws;
  unsigned short* Xb    = p; p += (size_t)S_LEN * DM;
  unsigned short* Wqkvt = p; p += (size_t)DQKV * DM;
  unsigned short* Wot   = p; p += (size_t)DM * DM;
  unsigned short* Qb    = p; p += (size_t)S_LEN * DM;
  unsigned short* Kb    = p; p += (size_t)S_LEN * DM;
  unsigned short* Vt    = p; p += (size_t)S_LEN * DM;
  unsigned short* Cb    = p;                              // 48 MB total

  cvt_kernel  <<<(S_LEN * DM) / (256 * 4), 256, 0, stream>>>(x, Xb);
  trans_kernel<<<dim3(DQKV / 32, DM / 32), dim3(32, 8), 0, stream>>>(Wqkv, Wqkvt, DM, DQKV);
  trans_kernel<<<dim3(DM / 32, DM / 32),   dim3(32, 8), 0, stream>>>(Wo,   Wot,   DM, DM);

  qkv_kernel <<<dim3(DQKV / 64, S_LEN / 64), 128, 0, stream>>>(Xb, Wqkvt, bqkv, Qb, Kb, Vt);
  attn_kernel<<<dim3(S_LEN / 16, NH),        32,  0, stream>>>(Qb, Kb, Vt, pad, Cb);
  proj_kernel<<<dim3(DM / 64, S_LEN / 64),   128, 0, stream>>>(Cb, Wot, bo, out);
}
